// AttentionLayer_17901423689910
// MI455X (gfx1250) — matrix operand order repaired and checked by reading, not compiled, not run on hardware
//
#include <hip/hip_runtime.h>
#include <hip/hip_bf16.h>
#include <math.h>

// ---------------------------------------------------------------------------
// Attention layer for MI455X (gfx1250, wave32, WMMA, async global->LDS).
//   Q = xWq+bq ; K = xWk+bk ; Vt = (xWv+bv)^T      (bf16 in d_ws)
//   S = QK^T/sqrt(D) ; P = softmax(S)              (f32 in d_out, bf16 in d_ws)
//   ctx = P V                                      (f32 in d_out)
// Matmuls: v_wmma_f32_16x16x32_bf16.  Big batched GEMMs use double-buffered
// global_load_async_to_lds_b128 + s_wait_asynccnt pipelining.
// ---------------------------------------------------------------------------

typedef __bf16 bf16_t;
typedef __attribute__((ext_vector_type(16))) __bf16 v16bf;
typedef __attribute__((ext_vector_type(8)))  __bf16 v8bf;
typedef __attribute__((ext_vector_type(8)))  float  v8f;

static constexpr int Bb   = 8;
static constexpr int Ss   = 2048;
static constexpr int Din  = 1024;
static constexpr int Datt = 1024;

static constexpr int BM = 128;
static constexpr int BN = 128;
static constexpr int BK = 32;
static constexpr int NT = 256;   // 8 wave32s

// 16-byte async copy: global -> LDS, bypasses VGPRs, tracked by ASYNCcnt.
__device__ __forceinline__ void async_cp16(unsigned lds_off, const void* gptr) {
    asm volatile("global_load_async_to_lds_b128 %0, %1, off"
                 :: "v"(lds_off), "v"(gptr) : "memory");
}
__device__ __forceinline__ void wait_async4() {
    asm volatile("s_wait_asynccnt %0" :: "n"(4) : "memory");
}
__device__ __forceinline__ void wait_async0() {
    asm volatile("s_wait_asynccnt %0" :: "n"(0) : "memory");
}

// ---------------------------------------------------------------------------
// Tiled WMMA GEMM:  C[M,N] = scale * (A[M,K] x B[K,N]) (+ bias[N])
//  A_F32   : A f32 in global (converted to bf16 on LDS fill), else bf16
//  B_F32   : B f32 in global, else bf16
//  B_TRANS : B element (k,n) read from Bp[n*ldb + k]  (k-contiguous)
//  OUT_BF16: store bf16 ; OUT_VT: store bf16 transposed as [B][N][Ss]
//  ASYNC   : requires (!A_F32 && !B_F32 && B_TRANS); double-buffered
//            global_load_async_to_lds_b128 pipeline.
// Assumes M,N divisible by 128 and K divisible by 32 (true for all uses).
// ---------------------------------------------------------------------------
template<bool A_F32, bool B_F32, bool B_TRANS, bool HAS_BIAS,
         bool OUT_BF16, bool OUT_VT, bool ASYNC>
__global__ __launch_bounds__(NT)
void wmma_gemm(const void* __restrict__ Ap, const void* __restrict__ Bp,
               const float* __restrict__ bias, void* __restrict__ Cp,
               int M, int N, int Kd, int lda, int ldb, int ldc,
               long long aBatch, long long bBatch, long long cBatch,
               float scale)
{
    __shared__ __align__(16) bf16_t As[2][BM][BK];   // [buf][m][k]
    __shared__ __align__(16) bf16_t Bs[2][BN][BK];   // [buf][n][k]

    const int tid  = threadIdx.x;
    const int lane = tid & 31;
    const int wave = tid >> 5;      // 0..7
    const int wm   = wave & 3;      // 4 waves along M -> 32 rows each
    const int wn   = wave >> 2;     // 2 waves along N -> 64 cols each

    const int m0 = blockIdx.y * BM;
    const int n0 = blockIdx.x * BN;
    const int bz = blockIdx.z;

    const float*  Af  = (const float*) Ap + (long long)bz * aBatch;
    const bf16_t* Ab  = (const bf16_t*)Ap + (long long)bz * aBatch;
    const float*  Bf  = (const float*) Bp + (long long)bz * bBatch;
    const bf16_t* Bbp = (const bf16_t*)Bp + (long long)bz * bBatch;
    const long long cOff = (long long)bz * cBatch;

    const v8f vzero = {0.f,0.f,0.f,0.f,0.f,0.f,0.f,0.f};
    v8f acc[2][4];
    #pragma unroll
    for (int i = 0; i < 2; ++i)
        #pragma unroll
        for (int j = 0; j < 4; ++j) acc[i][j] = vzero;

    // ---- LDS fragment loaders (ISA 16-bit A/B per-lane layouts) ----
    auto load_a = [&](int buf, int mbase) -> v16bf {
        const int row = mbase + (lane & 15);
        const int ko  = (lane >> 4) * 8;          // lanes 16-31: K=8..15 / 24..31
        v8bf lo = *(const v8bf*)&As[buf][row][ko];
        v8bf hi = *(const v8bf*)&As[buf][row][ko + 16];
        v16bf f;
        #pragma unroll
        for (int i = 0; i < 8; ++i) { f[i] = lo[i]; f[i + 8] = hi[i]; }
        return f;
    };
    auto load_b = [&](int buf, int nbase) -> v16bf {
        const int col = nbase + (lane & 15);
        const int kb  = (lane >> 4) * 8;          // lanes 16-31: K=8..15 / 24..31
        v8bf lo = *(const v8bf*)&Bs[buf][col][kb];
        v8bf hi = *(const v8bf*)&Bs[buf][col][kb + 16];
        v16bf f;
        #pragma unroll
        for (int i = 0; i < 8; ++i) { f[i] = lo[i]; f[i + 8] = hi[i]; }
        return f;
    };
    auto mma_step = [&](int buf) {
        v16bf a0 = load_a(buf, wm * 32);
        v16bf a1 = load_a(buf, wm * 32 + 16);
        #pragma unroll
        for (int j = 0; j < 4; ++j) {
            v16bf bf = load_b(buf, wn * 64 + j * 16);
            acc[0][j] = __builtin_amdgcn_wmma_f32_16x16x32_bf16(
                false, a0, false, bf, (short)0, acc[0][j], false, false);
            acc[1][j] = __builtin_amdgcn_wmma_f32_16x16x32_bf16(
                false, a1, false, bf, (short)0, acc[1][j], false, false);
        }
    };

    if (ASYNC) {
        // ---- double-buffered async global->LDS pipeline (bf16, B_TRANS) ----
        // Tile = 128 rows x 64 bytes = 512 16B-chunks; 256 threads x 2 chunks.
        auto fill_async = [&](int buf, int k0) {
            #pragma unroll
            for (int it = 0; it < 2; ++it) {
                const int c    = tid + it * NT;       // 0..511
                const int row  = c >> 2;
                const int boff = (c & 3) * 16;
                const char* ga = (const char*)(Ab + (long long)(m0 + row) * lda + k0) + boff;
                const char* gb = (const char*)(Bbp + (long long)(n0 + row) * ldb + k0) + boff;
                async_cp16((unsigned)(uintptr_t)&As[buf][row][0] + boff, ga);
                async_cp16((unsigned)(uintptr_t)&Bs[buf][row][0] + boff, gb);
            }
        };
        const int nk = Kd / BK;
        fill_async(0, 0);
        for (int ks = 0; ks < nk; ++ks) {
            const int cur = ks & 1;
            if (ks + 1 < nk) {
                fill_async(cur ^ 1, (ks + 1) * BK);
                wait_async4();                  // tile ks retired (in-order)
            } else {
                wait_async0();
            }
            __syncthreads();                    // all waves' tile ks in LDS
            mma_step(cur);
            __syncthreads();                    // buffer free for next issue
        }
    } else {
        // ---- synchronous fill with on-the-fly f32->bf16 conversion ----
        for (int k0 = 0; k0 < Kd; k0 += BK) {
            __syncthreads();
            {   // A tile: thread t -> row t/2, 16-element k-half
                const int r  = tid >> 1;
                const int kh = (tid & 1) * 16;
                const long long gr = (long long)(m0 + r);
                if (A_F32) {
                    const float4* s4 = (const float4*)(Af + gr * lda + (k0 + kh));
                    if (k0 + BK < Kd)
                        __builtin_prefetch((const char*)s4 + BK * sizeof(float), 0, 0);
                    #pragma unroll
                    for (int q = 0; q < 4; ++q) {
                        float4 v = s4[q];
                        As[0][r][kh + q*4 + 0] = (bf16_t)v.x;
                        As[0][r][kh + q*4 + 1] = (bf16_t)v.y;
                        As[0][r][kh + q*4 + 2] = (bf16_t)v.z;
                        As[0][r][kh + q*4 + 3] = (bf16_t)v.w;
                    }
                } else {
                    const bf16_t* s = Ab + gr * lda + (k0 + kh);
                    *(v8bf*)&As[0][r][kh]     = *(const v8bf*)(s);
                    *(v8bf*)&As[0][r][kh + 8] = *(const v8bf*)(s + 8);
                }
            }
            if (B_TRANS) {      // B(k,n) = Bp[n*ldb + k], k-contiguous
                const int n  = tid >> 1;
                const int kh = (tid & 1) * 16;
                const long long gn = (long long)(n0 + n);
                if (B_F32) {
                    const float4* s4 = (const float4*)(Bf + gn * ldb + (k0 + kh));
                    #pragma unroll
                    for (int q = 0; q < 4; ++q) {
                        float4 v = s4[q];
                        Bs[0][n][kh + q*4 + 0] = (bf16_t)v.x;
                        Bs[0][n][kh + q*4 + 1] = (bf16_t)v.y;
                        Bs[0][n][kh + q*4 + 2] = (bf16_t)v.z;
                        Bs[0][n][kh + q*4 + 3] = (bf16_t)v.w;
                    }
                } else {
                    const bf16_t* s = Bbp + gn * ldb + (k0 + kh);
                    *(v8bf*)&Bs[0][n][kh]     = *(const v8bf*)(s);
                    *(v8bf*)&Bs[0][n][kh + 8] = *(const v8bf*)(s + 8);
                }
            } else {            // B(k,n) = Bp[k*ldb + n], scatter into Bs[n][k]
                const int k  = tid >> 3;
                const int nh = (tid & 7) * 16;
                const long long gk = (long long)(k0 + k);
                if (B_F32) {
                    const float4* s4 = (const float4*)(Bf + gk * ldb + (n0 + nh));
                    if (k0 + BK < Kd)
                        __builtin_prefetch((const char*)s4 + (size_t)BK * ldb * sizeof(float), 0, 0);
                    #pragma unroll
                    for (int q = 0; q < 4; ++q) {
                        float4 v = s4[q];
                        Bs[0][nh + q*4 + 0][k] = (bf16_t)v.x;
                        Bs[0][nh + q*4 + 1][k] = (bf16_t)v.y;
                        Bs[0][nh + q*4 + 2][k] = (bf16_t)v.z;
                        Bs[0][nh + q*4 + 3][k] = (bf16_t)v.w;
                    }
                } else {
                    const bf16_t* s = Bbp + gk * ldb + (n0 + nh);
                    v8bf lo = *(const v8bf*)(s);
                    v8bf hi = *(const v8bf*)(s + 8);
                    #pragma unroll
                    for (int i = 0; i < 8; ++i) {
                        Bs[0][nh + i][k]     = lo[i];
                        Bs[0][nh + 8 + i][k] = hi[i];
                    }
                }
            }
            __syncthreads();
            mma_step(0);
        }
    }

    // ---- epilogue: C/D f32 layout: VGPR e -> M = e + (lane>=16)*8, N = lane&15
    #pragma unroll
    for (int mi = 0; mi < 2; ++mi) {
        #pragma unroll
        for (int j = 0; j < 4; ++j) {
            const int col = n0 + wn * 64 + j * 16 + (lane & 15);
            const float bv = HAS_BIAS ? bias[col] : 0.0f;
            #pragma unroll
            for (int e = 0; e < 8; ++e) {
                const int row = m0 + wm * 32 + mi * 16 + e + ((lane >> 4) * 8);
                const float v = acc[mi][j][e] * scale + bv;
                if (OUT_VT) {
                    // bf16, transposed per batch: [b][col][s], b=row/Ss, s=row%Ss
                    const long long idx =
                        ((long long)(row >> 11) * N + col) * (long long)Ss + (row & (Ss - 1));
                    ((bf16_t*)Cp)[idx] = (bf16_t)v;
                } else {
                    const long long idx = cOff + (long long)row * ldc + col;
                    if (OUT_BF16) ((bf16_t*)Cp)[idx] = (bf16_t)v;
                    else          ((float*)Cp)[idx]  = v;
                }
            }
        }
    }
}

// ---------------------------------------------------------------------------
// In-place row softmax over 2048 columns; also emits bf16 probabilities.
// One 256-thread block per row.
// ---------------------------------------------------------------------------
__global__ __launch_bounds__(NT)
void softmax_rows(float* __restrict__ scores, bf16_t* __restrict__ pbf)
{
    __shared__ float red[NT];
    const long long row = blockIdx.x;
    float*  p  = scores + row * (long long)Ss;
    bf16_t* pb = pbf    + row * (long long)Ss;
    const int tid = threadIdx.x;

    float vals[Ss / NT];               // 8 per thread
    float m = -INFINITY;
    #pragma unroll
    for (int i = 0; i < Ss / NT; ++i) {
        vals[i] = p[tid + i * NT];
        m = fmaxf(m, vals[i]);
    }
    red[tid] = m; __syncthreads();
    for (int s = NT / 2; s > 0; s >>= 1) {
        if (tid < s) red[tid] = fmaxf(red[tid], red[tid + s]);
        __syncthreads();
    }
    m = red[0]; __syncthreads();

    float sum = 0.f;
    #pragma unroll
    for (int i = 0; i < Ss / NT; ++i) {
        vals[i] = __expf(vals[i] - m);
        sum += vals[i];
    }
    red[tid] = sum; __syncthreads();
    for (int s = NT / 2; s > 0; s >>= 1) {
        if (tid < s) red[tid] += red[tid + s];
        __syncthreads();
    }
    const float inv = 1.0f / red[0];
    #pragma unroll
    for (int i = 0; i < Ss / NT; ++i) {
        const float w = vals[i] * inv;
        p [tid + i * NT] = w;
        pb[tid + i * NT] = (bf16_t)w;
    }
}

// ---------------------------------------------------------------------------
extern "C" void kernel_launch(void* const* d_in, const int* in_sizes, int n_in,
                              void* d_out, int out_size, void* d_ws, size_t ws_size,
                              hipStream_t stream)
{
    (void)in_sizes; (void)n_in; (void)out_size; (void)ws_size;

    const float* x  = (const float*)d_in[0];
    const float* Wq = (const float*)d_in[1];
    const float* bq = (const float*)d_in[2];
    const float* Wk = (const float*)d_in[3];
    const float* bk = (const float*)d_in[4];
    const float* Wv = (const float*)d_in[5];
    const float* bv = (const float*)d_in[6];

    float* ctx  = (float*)d_out;                               // [B,S,Datt]
    float* attn = ctx + (long long)Bb * Ss * Datt;             // [B,S,S]

    bf16_t* Qb = (bf16_t*)d_ws;                                // [B*S, Datt]
    bf16_t* Kb = Qb + (long long)Bb * Ss * Datt;               // [B*S, Datt]
    bf16_t* Vt = Kb + (long long)Bb * Ss * Datt;               // [B][Datt][Ss]
    bf16_t* Pb = Vt + (long long)Bb * Ss * Datt;               // [B][S][S]

    const dim3 blk(NT);
    const int Mqkv = Bb * Ss;                                  // 16384

    // --- Stage 1: Q/K projections (f32 -> bf16, +bias); V stored transposed ---
    {
        dim3 grid(Datt / BN, Mqkv / BM, 1);
        wmma_gemm<true, true, false, true, true,  false, false><<<grid, blk, 0, stream>>>(
            x, Wq, bq, Qb, Mqkv, Datt, Din, Din, Datt, Datt, 0, 0, 0, 1.0f);
        wmma_gemm<true, true, false, true, true,  false, false><<<grid, blk, 0, stream>>>(
            x, Wk, bk, Kb, Mqkv, Datt, Din, Din, Datt, Datt, 0, 0, 0, 1.0f);
        wmma_gemm<true, true, false, true, false, true,  false><<<grid, blk, 0, stream>>>(
            x, Wv, bv, Vt, Mqkv, Datt, Din, Din, Datt, Datt, 0, 0, 0, 1.0f);
    }

    // --- Stage 2: scores = Q K^T / sqrt(Datt)  (async bf16 pipeline) ---
    {
        dim3 grid(Ss / BN, Ss / BM, Bb);
        wmma_gemm<false, false, true, false, false, false, true><<<grid, blk, 0, stream>>>(
            Qb, Kb, nullptr, attn, Ss, Ss, Datt, Datt, Datt, Ss,
            (long long)Ss * Datt, (long long)Ss * Datt, (long long)Ss * Ss,
            0.03125f /* 1/sqrt(1024) */);
    }

    // --- Stage 3: row softmax (f32 in place, bf16 copy for next GEMM) ---
    softmax_rows<<<dim3(Bb * Ss), blk, 0, stream>>>(attn, Pb);

    // --- Stage 4: context = P V  (async bf16 pipeline; B = V^T, B_TRANS) ---
    {
        dim3 grid(Datt / BN, Ss / BM, Bb);
        wmma_gemm<false, false, true, false, false, false, true><<<grid, blk, 0, stream>>>(
            Pb, Vt, nullptr, ctx, Ss, Datt, Ss, Ss, Ss, Datt,
            (long long)Ss * Ss, (long long)Datt * Ss, (long long)Ss * Datt,
            1.0f);
    }
}
